// DensePoseFCHead_37778532336316
// MI455X (gfx1250) — matrix operand order repaired and checked by reading, not compiled, not run on hardware
//
#include <hip/hip_runtime.h>
#include <hip/hip_bf16.h>

// ---------------------------------------------------------------------------
// DensePose FC head for MI455X (gfx1250, wave32, WMMA + TDM).
//   conv1 -> relu -> conv2 -> relu -> (identity)
//   pooled stats (top4-softmax mean, local-linear mean) -> attention (N,3)
//   dyn conv1 (att-combined weights) -> bn1 -> relu
//   dyn conv2 -> bn2 -> +identity -> relu
// Convs: implicit GEMM on v_wmma_f32_16x16x32_bf16, flattened K = c*9+tap,
// zero-padded bf16 activation planes (30x30), bf16 pre-converted static
// weights staged into LDS by the Tensor Data Mover.
// ---------------------------------------------------------------------------

#define NB     32
#define HID    512
#define HP     28
#define PIX    (HP * HP)           // 784
#define PH     30                  // padded plane side
#define PPLANE (PH * PH)           // 900
#define NPATCH 25
#define HS     112
#define NPIX   (HS * HS)           // 12544
#define KDYN   3
#define ATTH   64
#define BNEPS  1e-5f
#define KTOT   (HID * 9)           // 4608 flattened K
#define W1OFF  (HID * HID * 9)     // per-k dyn weight stride: 2359296

typedef __attribute__((ext_vector_type(8)))  float    v8f;
typedef __attribute__((ext_vector_type(16))) __bf16   v16bf;
typedef __attribute__((ext_vector_type(8)))  __bf16   v8bf;
typedef __attribute__((ext_vector_type(4)))  __bf16   v4bf;
typedef __attribute__((ext_vector_type(4)))  unsigned uv4;
typedef __attribute__((ext_vector_type(8)))  int      iv8;
typedef __attribute__((ext_vector_type(4)))  int      iv4;

#if defined(__has_builtin)
#if __has_builtin(__builtin_amdgcn_tensor_load_to_lds) && \
    __has_builtin(__builtin_amdgcn_s_wait_tensorcnt)
#define USE_TDM 1
#endif
#endif

static __device__ __forceinline__ __bf16 f2bf(float f) {
    // round-to-nearest-even f32 -> bf16
    unsigned u = __builtin_bit_cast(unsigned, f);
    unsigned r = (u + 0x7FFFu + ((u >> 16) & 1u)) >> 16;
    unsigned short s = (unsigned short)r;
    return __builtin_bit_cast(__bf16, s);
}
static __device__ __forceinline__ float bf2f(__bf16 h) {
    unsigned u = (unsigned)__builtin_bit_cast(unsigned short, h) << 16;
    return __builtin_bit_cast(float, u);
}

// ---------------------------------------------------------------------------
// Zero the padded bf16 buffers (halo must be 0; ws is poisoned to 0xAA).
// ---------------------------------------------------------------------------
__global__ void zero_kernel(uint4* __restrict__ p, unsigned n16) {
    unsigned i = blockIdx.x * 256u + threadIdx.x;
    const unsigned stride = gridDim.x * 256u;
    uint4 z; z.x = 0u; z.y = 0u; z.z = 0u; z.w = 0u;
    for (; i < n16; i += stride) p[i] = z;
}

// features f32 (N,512,28,28) -> padded bf16 planes (N*512, 30, 30) interior
__global__ void cvt_features_kernel(const float* __restrict__ in,
                                    __bf16* __restrict__ out) {
    const unsigned total = NB * HID * PIX;
    unsigned i = blockIdx.x * 256u + threadIdx.x;
    const unsigned stride = gridDim.x * 256u;
    for (; i < total; i += stride) {
        unsigned bc = i / PIX, p = i - bc * PIX;
        unsigned h = p / HP, w = p - h * HP;
        out[(size_t)bc * PPLANE + (h + 1) * PH + (w + 1)] = f2bf(in[i]);
    }
}

// static conv weights f32 -> bf16, 4 at a time
__global__ void cvt_weights_kernel(const float* __restrict__ in,
                                   __bf16* __restrict__ out, unsigned n4) {
    unsigned i = blockIdx.x * 256u + threadIdx.x;
    const unsigned stride = gridDim.x * 256u;
    for (; i < n4; i += stride) {
        float4 x = ((const float4*)in)[i];
        v4bf o; o[0] = f2bf(x.x); o[1] = f2bf(x.y); o[2] = f2bf(x.z); o[3] = f2bf(x.w);
        ((v4bf*)out)[i] = o;
    }
}

// ---------------------------------------------------------------------------
// Kernel: per-pixel softmax top-4 over 25 channels, pooled; argmax map.
// ---------------------------------------------------------------------------
__global__ void topk_stats_kernel(const float* __restrict__ fs,
                                  float* __restrict__ pooled1,
                                  int* __restrict__ amax) {
    const int b = blockIdx.x, tid = threadIdx.x;
    float s0 = 0.f, s1 = 0.f, s2 = 0.f, s3 = 0.f;
    const float* base = fs + (size_t)b * NPATCH * NPIX;
    for (int p = tid; p < NPIX; p += 256) {
        float fv[NPATCH];
        float t0 = -3.0e38f, t1 = t0, t2 = t0, t3 = t0;
        int am = 0;
        #pragma unroll
        for (int c = 0; c < NPATCH; ++c) {
            float f = base[c * NPIX + p];
            fv[c] = f;
            if (f > t0)      { t3 = t2; t2 = t1; t1 = t0; t0 = f; am = c; }
            else if (f > t1) { t3 = t2; t2 = t1; t1 = f; }
            else if (f > t2) { t3 = t2; t2 = f; }
            else if (f > t3) { t3 = f; }
        }
        float den = 0.f;
        #pragma unroll
        for (int c = 0; c < NPATCH; ++c) den += __expf(fv[c] - t0);
        float inv = 1.f / den;
        s0 += inv;
        s1 += __expf(t1 - t0) * inv;
        s2 += __expf(t2 - t0) * inv;
        s3 += __expf(t3 - t0) * inv;
        amax[b * NPIX + p] = am;
    }
    __shared__ float red[256];
    float sums[5] = { s0, s1, s2, s3, (s0 + s1 + s2 + s3) * 0.25f };
    for (int j = 0; j < 5; ++j) {
        red[tid] = sums[j];
        __syncthreads();
        for (int s = 128; s > 0; s >>= 1) {
            if (tid < s) red[tid] += red[tid + s];
            __syncthreads();
        }
        if (tid == 0) pooled1[b * 5 + j] = red[0] / (float)NPIX;
        __syncthreads();
    }
}

// ---------------------------------------------------------------------------
// Kernel: local-linear statistic, spatially pooled. One block per sample.
// ---------------------------------------------------------------------------
__global__ void local_stats_kernel(const float* __restrict__ u,
                                   const float* __restrict__ v,
                                   const int* __restrict__ amax,
                                   float* __restrict__ pooled2) {
    const int b = blockIdx.x, tid = threadIdx.x;
    const int CH = NPATCH - 1;            // 24
    const int CL = CH * HS;               // 2688
    extern __shared__ float sm[];
    float* cntH = sm;
    float* accU = cntH + CL;
    float* cntW = accU + CL;
    float* accV = cntW + CL;
    float* sqU  = accV + CL;
    float* sqV  = sqU  + CL;
    for (int i = tid; i < 6 * CL; i += 256) sm[i] = 0.f;
    __syncthreads();

    for (int p = tid; p < NPIX; p += 256) {
        int c = amax[b * NPIX + p];
        if (c >= 1) {
            int ch = c - 1, h = p / HS, w = p % HS;
            float uu = u[((size_t)b * NPATCH + c) * NPIX + p];
            float vv = v[((size_t)b * NPATCH + c) * NPIX + p];
            atomicAdd(&cntH[ch * HS + h], 1.f);
            atomicAdd(&accU[ch * HS + h], uu);
            atomicAdd(&cntW[ch * HS + w], 1.f);
            atomicAdd(&accV[ch * HS + w], vv);
        }
    }
    __syncthreads();
    for (int i = tid; i < CL; i += 256) {
        float ch = cntH[i]; accU[i] = (ch > 0.f) ? accU[i] / ch : 0.f;
        float cw = cntW[i]; accV[i] = (cw > 0.f) ? accV[i] / cw : 0.f;
    }
    __syncthreads();
    for (int p = tid; p < NPIX; p += 256) {
        int c = amax[b * NPIX + p];
        if (c >= 1) {
            int ch = c - 1, h = p / HS, w = p % HS;
            float uu = u[((size_t)b * NPATCH + c) * NPIX + p];
            float vv = v[((size_t)b * NPATCH + c) * NPIX + p];
            float du = uu - accU[ch * HS + h];
            float dv = vv - accV[ch * HS + w];
            atomicAdd(&sqU[ch * HS + h], du * du);
            atomicAdd(&sqV[ch * HS + w], dv * dv);
        }
    }
    __syncthreads();
    for (int i = tid; i < CL; i += 256) {
        float ch = cntH[i]; accU[i] = (ch > 0.f) ? sqrtf(sqrtf(sqU[i] / ch)) : 0.f;
        float cw = cntW[i]; accV[i] = (cw > 0.f) ? sqrtf(sqrtf(sqV[i] / cw)) : 0.f;
    }
    __syncthreads();
    float acc = 0.f;
    for (int p = tid; p < NPIX; p += 256) {
        int h = p / HS, w = p % HS;
        float S = 0.f;
        #pragma unroll 8
        for (int ch = 0; ch < CH; ++ch) S += accU[ch * HS + h] * accV[ch * HS + w];
        acc += __expf(-S);
    }
    __shared__ float red[256];
    red[tid] = acc;
    __syncthreads();
    for (int s = 128; s > 0; s >>= 1) {
        if (tid < s) red[tid] += red[tid + s];
        __syncthreads();
    }
    if (tid == 0) pooled2[b] = red[0] / (float)NPIX;
}

// ---------------------------------------------------------------------------
// Kernel: both attention heads. One block per sample, 64 threads.
// ---------------------------------------------------------------------------
__global__ void att_kernel(const float* __restrict__ p1, const float* __restrict__ p2,
                           const float* __restrict__ f1w, const float* __restrict__ f1b,
                           const float* __restrict__ f2w, const float* __restrict__ f2b,
                           const float* __restrict__ g1w, const float* __restrict__ g1b,
                           const float* __restrict__ g2w, const float* __restrict__ g2b,
                           float* __restrict__ att1, float* __restrict__ att2) {
    const int b = blockIdx.x, j = threadIdx.x;
    __shared__ float h1[ATTH], h2[ATTH], l1[KDYN], l2[KDYN];
    float a = f1b[j];
    #pragma unroll
    for (int i = 0; i < 5; ++i) a += f1w[j * 5 + i] * p1[b * 5 + i];
    h1[j] = fmaxf(a, 0.f);
    h2[j] = fmaxf(g1b[j] + g1w[j] * p2[b], 0.f);
    __syncthreads();
    if (j < KDYN) {
        float s = f2b[j], t = g2b[j];
        #pragma unroll 8
        for (int i = 0; i < ATTH; ++i) { s += f2w[j * ATTH + i] * h1[i];
                                         t += g2w[j * ATTH + i] * h2[i]; }
        l1[j] = s; l2[j] = t;
    }
    __syncthreads();
    if (j == 0) {
        float m1 = fmaxf(l1[0], fmaxf(l1[1], l1[2]));
        float e0 = __expf(l1[0] - m1), e1 = __expf(l1[1] - m1), e2 = __expf(l1[2] - m1);
        float s = e0 + e1 + e2;
        att1[b * 3 + 0] = e0 / s; att1[b * 3 + 1] = e1 / s; att1[b * 3 + 2] = e2 / s;
        float m2 = fmaxf(l2[0], fmaxf(l2[1], l2[2]));
        float f0 = __expf(l2[0] - m2), f1 = __expf(l2[1] - m2), f2 = __expf(l2[2] - m2);
        float q = f0 + f1 + f2;
        att2[b * 3 + 0] = f0 / q; att2[b * 3 + 1] = f1 / q; att2[b * 3 + 2] = f2 / q;
    }
}

// ---------------------------------------------------------------------------
// Implicit-GEMM 3x3 SAME conv via v_wmma_f32_16x16x32_bf16.
// Grid (13 pixel-tiles, 8 m-tiles, 32 samples), 256 threads = 8 wave32.
// MODE 0/1: bf16 weights (TDM-staged A tile) + bias/relu -> padded bf16 out
// MODE 2:   f32 att-mixed weights, bn1/relu -> padded bf16 out
// MODE 3:   f32 att-mixed weights, bn2 + residual + relu -> f32 d_out
// B staging is fully branch-free: out-of-range columns load from a clamped
// address (their GEMM columns are never stored).
// ---------------------------------------------------------------------------
template <int MODE>
__global__ void conv_gemm_kernel(const __bf16* __restrict__ inpb,
                                 const __bf16* __restrict__ wgtb,  // MODE <= 1
                                 const float*  __restrict__ wgtf,  // MODE >= 2
                                 const float*  __restrict__ att,
                                 const float*  __restrict__ bias,
                                 const float*  __restrict__ bng,
                                 const float*  __restrict__ bnb,
                                 const float*  __restrict__ bnm,
                                 const float*  __restrict__ bnv,
                                 const __bf16* __restrict__ residb,
                                 __bf16*       __restrict__ outb,
                                 float*        __restrict__ outf) {
    const int b      = blockIdx.z;
    const int n_base = blockIdx.x * 64;
    const int m_base = blockIdx.y * 64;
    const int tid    = threadIdx.x;
    const int lane   = tid & 31;
    const int wave   = tid >> 5;
    const int msub   = wave >> 1;
    const int npair  = wave & 1;

    float a0 = 1.f, a1 = 0.f, a2 = 0.f;
    if (MODE >= 2) { a0 = att[b * 3 + 0]; a1 = att[b * 3 + 1]; a2 = att[b * 3 + 2]; }

    __shared__ __align__(16) __bf16 As[64 * 40];   // [m][k], padded stride 40
    __shared__ __align__(16) __bf16 Bs[64 * 40];   // [n][k], padded stride 40

    v8f acc0 = {}; v8f acc1 = {};

    const __bf16* inB = inpb + (size_t)b * HID * PPLANE;

    // A staging (manual / dyn): thread t -> row m = t>>2, k run k0 = (t&3)*8
    const int am_ = tid >> 2;
    const int ak0 = (tid & 3) * 8;

    // B staging: thread t -> col n = t&63, contiguous k run (t>>6)*8 + j.
    // Clamp out-of-range pixels to a valid address (columns never stored).
    const int bn_ = tid & 63;
    const int bk8 = (tid >> 6) * 8;
    const int bp  = n_base + bn_;
    const int bpc = (bp < PIX) ? bp : (PIX - 1);
    const int bh  = bpc / HP, bw = bpc % HP;
    const int ppad = (bh + 1) * PH + (bw + 1);            // interior padded idx

    for (int kbase = 0; kbase < KTOT; kbase += 32) {
        // ---- stage A tile ------------------------------------------------
        if (MODE <= 1) {
#ifdef USE_TDM
            if (wave == 0) {
                // 2D TDM tile: 32 (k) x 64 (m) bf16, row stride 4608 elems,
                // LDS pad +4 DWORDs every 16 DWORDs to match stride-40 rows.
                unsigned ldsa = (unsigned)(size_t)(void*)&As[0];
                unsigned long long ga =
                    (unsigned long long)(size_t)(const void*)(wgtb +
                        (size_t)m_base * KTOT + kbase);
                uv4 g0 = { 1u, ldsa, (unsigned)ga,
                           ((unsigned)(ga >> 32) & 0x01FFFFFFu) | 0x80000000u };
                iv8 g1 = { 0x06D10000,      // data_size=2B, pad en, int=16dw, amt=4dw
                           0x12000000,      // tensor_dim0 = 4608 (lo16 << 16)
                           0x02000000,      // tensor_dim1 = 512  (lo16 << 16)
                           0x00200000,      // tile_dim0 = 32 (<<16)
                           64,              // tile_dim1 = 64
                           4608,            // tensor_dim0_stride
                           0, 0 };
                iv4 gz4 = { 0, 0, 0, 0 };
                iv8 gz8 = { 0, 0, 0, 0, 0, 0, 0, 0 };
                __builtin_amdgcn_tensor_load_to_lds(g0, g1, gz4, gz4, gz8, 0);
            }
#else
            {
                const __bf16* wp = wgtb + (size_t)(m_base + am_) * KTOT + kbase + ak0;
                *(v8bf*)&As[am_ * 40 + ak0] = *(const v8bf*)wp;
            }
#endif
        } else {
            const float* wp = wgtf + (size_t)(m_base + am_) * KTOT + kbase + ak0;
            float4 xa = *(const float4*)wp;
            float4 xb = *(const float4*)(wp + 4);
            float4 ya = *(const float4*)(wp + W1OFF);
            float4 yb = *(const float4*)(wp + W1OFF + 4);
            float4 za = *(const float4*)(wp + 2 * W1OFF);
            float4 zb = *(const float4*)(wp + 2 * W1OFF + 4);
            float wv[8] = { xa.x, xa.y, xa.z, xa.w, xb.x, xb.y, xb.z, xb.w };
            float yv[8] = { ya.x, ya.y, ya.z, ya.w, yb.x, yb.y, yb.z, yb.w };
            float zv[8] = { za.x, za.y, za.z, za.w, zb.x, zb.y, zb.z, zb.w };
            v8bf ap;
            #pragma unroll
            for (int i = 0; i < 8; ++i)
                ap[i] = f2bf(a0 * wv[i] + a1 * yv[i] + a2 * zv[i]);
            *(v8bf*)&As[am_ * 40 + ak0] = ap;
            if (kbase + 32 < KTOT) __builtin_prefetch(wp + 32, 0, 3);
        }

        // ---- stage B tile: branch-free gather, one b128 LDS store --------
        {
            const int kk = kbase + bk8;
            const int c0 = kk / 9;            // one magic-div per iteration
            const int t0 = kk - 9 * c0;       // 0..8
            __bf16 xv[8];
            #pragma unroll
            for (int j = 0; j < 8; ++j) {
                int t  = t0 + j;                              // 0..15
                int wr = (t >= 9) ? 1 : 0;                    // wraps at most once
                int tm = t - 9 * wr;                          // tap 0..8
                int dr = (tm >= 6) ? 2 : ((tm >= 3) ? 1 : 0); // tap row
                int sh = dr * PH + (tm - 3 * dr) - (PH + 1);  // plane offset
                xv[j] = inB[(c0 + wr) * PPLANE + ppad + sh];
            }
            v8bf bpk;
            #pragma unroll
            for (int j = 0; j < 8; ++j) bpk[j] = xv[j];
            *(v8bf*)&Bs[bn_ * 40 + bk8] = bpk;
        }

#ifdef USE_TDM
        if (MODE <= 1 && wave == 0) __builtin_amdgcn_s_wait_tensorcnt(0);
#endif
        __syncthreads();

        // ---- A fragment: row m, K runs {k0..k0+7, k0+16..k0+23}
        const int mrow = msub * 16 + (lane & 15);
        const int k0   = (lane < 16) ? 0 : 8;
        v8bf alo = *(const v8bf*)&As[mrow * 40 + k0];
        v8bf ahi = *(const v8bf*)&As[mrow * 40 + k0 + 16];
        v16bf afr;
        #pragma unroll
        for (int i = 0; i < 8; ++i) { afr[i] = alo[i]; afr[8 + i] = ahi[i]; }

        // ---- two B fragments + WMMA
        #pragma unroll
        for (int t = 0; t < 2; ++t) {
            const int nsub = npair * 2 + t;
            const int ncol = nsub * 16 + (lane & 15);
            const int kh   = (lane < 16) ? 0 : 8;
            v8bf blo = *(const v8bf*)&Bs[ncol * 40 + kh];
            v8bf bhi = *(const v8bf*)&Bs[ncol * 40 + kh + 16];
            v16bf bfr;
            #pragma unroll
            for (int i = 0; i < 8; ++i) { bfr[i] = blo[i]; bfr[8 + i] = bhi[i]; }
            if (t == 0)
                acc0 = __builtin_amdgcn_wmma_f32_16x16x32_bf16(
                    false, afr, false, bfr, (short)0, acc0, false, false);
            else
                acc1 = __builtin_amdgcn_wmma_f32_16x16x32_bf16(
                    false, afr, false, bfr, (short)0, acc1, false, false);
        }
        __syncthreads();
    }

    // ---- fused epilogue + store (C layout: vgpr r, lane<16 -> M=r, else M=r+8)
    #pragma unroll
    for (int t = 0; t < 2; ++t) {
        v8f accv = (t == 0) ? acc0 : acc1;
        const int nsub = npair * 2 + t;
        const int p = n_base + nsub * 16 + (lane & 15);
        if (p < PIX) {
            const int h = p / HP, w = p % HP;
            const int pp = (h + 1) * PH + (w + 1);
            #pragma unroll
            for (int r = 0; r < 8; ++r) {
                int m = m_base + msub * 16 + ((lane < 16) ? r : r + 8);
                float vv = accv[r];
                if (MODE <= 1) {
                    vv = fmaxf(vv + bias[m], 0.f);
                    outb[((size_t)b * HID + m) * PPLANE + pp] = f2bf(vv);
                } else if (MODE == 2) {
                    float inv = bng[m] * rsqrtf(bnv[m] + BNEPS);
                    vv = fmaxf((vv - bnm[m]) * inv + bnb[m], 0.f);
                    outb[((size_t)b * HID + m) * PPLANE + pp] = f2bf(vv);
                } else {
                    float inv = bng[m] * rsqrtf(bnv[m] + BNEPS);
                    vv = (vv - bnm[m]) * inv + bnb[m];
                    vv += bf2f(residb[((size_t)b * HID + m) * PPLANE + pp]);
                    outf[((size_t)b * HID + m) * PIX + p] = fmaxf(vv, 0.f);
                }
            }
        }
    }
}

// ---------------------------------------------------------------------------
extern "C" void kernel_launch(void* const* d_in, const int* in_sizes, int n_in,
                              void* d_out, int out_size, void* d_ws, size_t ws_size,
                              hipStream_t stream) {
    const float* features   = (const float*)d_in[0];
    const float* fine_segm  = (const float*)d_in[1];
    const float* u          = (const float*)d_in[2];
    const float* v          = (const float*)d_in[3];
    const float* conv1_w    = (const float*)d_in[4];
    const float* conv1_b    = (const float*)d_in[5];
    const float* conv2_w    = (const float*)d_in[6];
    const float* conv2_b    = (const float*)d_in[7];
    const float* att1_fc1_w = (const float*)d_in[8];
    const float* att1_fc1_b = (const float*)d_in[9];
    const float* att1_fc2_w = (const float*)d_in[10];
    const float* att1_fc2_b = (const float*)d_in[11];
    const float* dyn1_w     = (const float*)d_in[12];
    const float* att2_fc1_w = (const float*)d_in[13];
    const float* att2_fc1_b = (const float*)d_in[14];
    const float* att2_fc2_w = (const float*)d_in[15];
    const float* att2_fc2_b = (const float*)d_in[16];
    const float* dyn2_w     = (const float*)d_in[17];
    const float* bn1_g = (const float*)d_in[18];
    const float* bn1_b = (const float*)d_in[19];
    const float* bn1_m = (const float*)d_in[20];
    const float* bn1_v = (const float*)d_in[21];
    const float* bn2_g = (const float*)d_in[22];
    const float* bn2_b = (const float*)d_in[23];
    const float* bn2_m = (const float*)d_in[24];
    const float* bn2_v = (const float*)d_in[25];
    float* outp = (float*)d_out;

    // workspace: 4 padded bf16 plane buffers + 2 bf16 weight buffers + small
    const size_t PBUF = (size_t)NB * HID * PPLANE;    // 14,745,600 bf16 each
    const size_t WSZ  = (size_t)HID * KTOT;           // 2,359,296 bf16 each
    __bf16* fbuf = (__bf16*)d_ws;
    __bf16* xbuf = fbuf + PBUF;
    __bf16* ybuf = xbuf + PBUF;
    __bf16* zbuf = ybuf + PBUF;
    __bf16* wbf1 = zbuf + PBUF;
    __bf16* wbf2 = wbf1 + WSZ;
    float* pooled1 = (float*)(wbf2 + WSZ);
    float* pooled2 = pooled1 + NB * 5;
    float* att1    = pooled2 + NB;
    float* att2    = att1 + NB * 3;
    int*   amax    = (int*)(att2 + NB * 3);           // NB * NPIX ints

    // stage 0: zero padded buffers, convert features + static weights to bf16
    const unsigned n16 = (unsigned)(4 * PBUF * sizeof(__bf16) / 16);
    zero_kernel<<<4096, 256, 0, stream>>>((uint4*)d_ws, n16);
    cvt_features_kernel<<<8192, 256, 0, stream>>>(features, fbuf);
    cvt_weights_kernel<<<2304, 256, 0, stream>>>(conv1_w, wbf1, (unsigned)(WSZ / 4));
    cvt_weights_kernel<<<2304, 256, 0, stream>>>(conv2_w, wbf2, (unsigned)(WSZ / 4));

    // stage 1: pooled segmentation statistics
    topk_stats_kernel<<<NB, 256, 0, stream>>>(fine_segm, pooled1, amax);
    const size_t smem2 = (size_t)6 * (NPATCH - 1) * HS * sizeof(float); // 64512 B
    local_stats_kernel<<<NB, 256, smem2, stream>>>(u, v, amax, pooled2);

    // stage 2: attention coefficients
    att_kernel<<<NB, ATTH, 0, stream>>>(pooled1, pooled2,
                                        att1_fc1_w, att1_fc1_b, att1_fc2_w, att1_fc2_b,
                                        att2_fc1_w, att2_fc1_b, att2_fc2_w, att2_fc2_b,
                                        att1, att2);

    // stage 3: four WMMA implicit-GEMM convs
    dim3 grid((PIX + 63) / 64, HID / 64, NB);         // (13, 8, 32)
    dim3 block(256);
    conv_gemm_kernel<0><<<grid, block, 0, stream>>>(
        fbuf, wbf1, nullptr, nullptr, conv1_b,
        nullptr, nullptr, nullptr, nullptr, nullptr, xbuf, nullptr);
    conv_gemm_kernel<1><<<grid, block, 0, stream>>>(
        xbuf, wbf2, nullptr, nullptr, conv2_b,
        nullptr, nullptr, nullptr, nullptr, nullptr, ybuf, nullptr);
    conv_gemm_kernel<2><<<grid, block, 0, stream>>>(
        ybuf, nullptr, dyn1_w, att1, nullptr,
        bn1_g, bn1_b, bn1_m, bn1_v, nullptr, zbuf, nullptr);
    conv_gemm_kernel<3><<<grid, block, 0, stream>>>(
        zbuf, nullptr, dyn2_w, att2, nullptr,
        bn2_g, bn2_b, bn2_m, bn2_v, ybuf, nullptr, outp);
}
